// TransFeature_71270687310399
// MI455X (gfx1250) — matrix operand order repaired and checked by reading, not compiled, not run on hardware
//
#include <hip/hip_runtime.h>
#include <hip/hip_bf16.h>
#include <math.h>

// ---------------------------------------------------------------------------
// Problem constants (from reference)
// ---------------------------------------------------------------------------
#define Bq    2
#define DD    8
#define HHs   128
#define WWs   128
#define CCs   64
#define NNs   (DD*HHs*WWs)     // 131072 tokens per batch
#define NHEAD 4
#define DHd   16

typedef __attribute__((ext_vector_type(16))) __bf16        v16bf;
typedef __attribute__((ext_vector_type(8)))  float         v8f;
typedef __attribute__((ext_vector_type(4)))  unsigned int  v4u;
typedef __attribute__((ext_vector_type(8)))  int           v8i;
typedef __attribute__((ext_vector_type(4)))  int           v4i;
typedef int v4i_vec __attribute__((vector_size(16)));      // b128 payload type

// ---------------------------------------------------------------------------
// Kernel 1: fused QKV GEMM (bf16 WMMA, f32 accum) + V store + attention
// statistics. q/k never touch HBM: Skq (16x16/head) is computed with WMMA
// (A = K^T from LDS, B = Q from LDS), accumulator carried in the C operand
// across slabs; Sq2/Sk2 diagonals are cheap scalar sums.
// grid: 256 blocks (128 per batch), 256 threads (8 waves), 1024 rows/block.
// ---------------------------------------------------------------------------
__global__ __launch_bounds__(256)
void k_qkv(const float* __restrict__ x,
           const float* __restrict__ Wq, const float* __restrict__ Wk,
           const float* __restrict__ Wv,
           float* __restrict__ vbuf, float* __restrict__ part)
{
    __shared__ __bf16 Wt[3 * 64 * 64];   // [mat][n][k] transposed weights, 24KB
    __shared__ __bf16 qs[128 * 64];      // q slab (bf16), 16KB
    __shared__ __bf16 ks[128 * 64];      // k slab (bf16), 16KB

    const int tid = threadIdx.x;
    const int wv  = tid >> 5;
    const int L   = tid & 31;
    const int b   = blockIdx.x >> 7;        // 128 blocks per batch
    const int blk = blockIdx.x & 127;
    const size_t batchBase = (size_t)b * NNs;
    const int rowBlockBase = blk * 1024;

    // Stage Wq|Wk|Wv transposed to [n][k]: B-fragment = 16 contiguous bf16.
    for (int i = tid; i < 3 * 64 * 64; i += 256) {
        int mat = i >> 12; int r = i & 4095; int n = r >> 6; int k = r & 63;
        const float* Wsrc = (mat == 0) ? Wq : (mat == 1) ? Wk : Wv;
        Wt[i] = (__bf16)Wsrc[k * 64 + n];
    }
    __syncthreads();

    // ISA A/B lane mappings (CDNA5 ISA 7.12.2, 16-bit 16x32 A / 32x16 B)
    const int Mrow = L & 15;
    const int hi   = (L >> 4) & 1;
    const int nL   = L & 15;

    // Wave -> (head, row-half) assignment for the Skq statistics GEMM
    const int sh    = wv & 3;     // head handled by this wave
    const int shalf = wv >> 2;    // which 64-row half of the slab

    v8f  skqAcc = {};             // persistent 16x16 accum (d x e) for (sh,shalf)
    float snorm = 0.f;            // tid<64: Sq2 entry; 64..127: Sk2 entry

    for (int slab = 0; slab < 8; ++slab) {
        const int waveRow = rowBlockBase + slab * 128 + wv * 16;
        const size_t gRow = batchBase + waveRow + Mrow;
        const float* xr = x + gRow * 64;

        if (slab < 7)  // stream-ahead hint -> global_prefetch_b8
            __builtin_prefetch(x + (gRow + 128) * 64, 0, 1);

        // A fragments: lane L owns row M=L&15; K(t) = 16*hi + t (contiguous order, same as B).
        v16bf a0, a1;
#pragma unroll
        for (int t = 0; t < 16; ++t) {
            int k0 = (hi << 4) + t;   // 0..31
            a0[t] = (__bf16)xr[k0];
            a1[t] = (__bf16)xr[32 + k0];
        }

#pragma unroll
        for (int mat = 0; mat < 3; ++mat) {
#pragma unroll
            for (int nt = 0; nt < 4; ++nt) {
                const __bf16* bp = &Wt[mat * 4096 + (nt * 16 + nL) * 64 + hi * 16];
                v16bf b0, b1;
#pragma unroll
                for (int t = 0; t < 16; ++t) { b0[t] = bp[t]; b1[t] = bp[32 + t]; }
                v8f acc = {};
                acc = __builtin_amdgcn_wmma_f32_16x16x32_bf16(
                          false, a0, false, b0, (short)0, acc, false, false);
                acc = __builtin_amdgcn_wmma_f32_16x16x32_bf16(
                          false, a1, false, b1, (short)0, acc, false, false);
                // D layout: VGPR r -> row (r + hi*8), col nt*16+nL
                if (mat == 2) {
                    float* vr = vbuf + (batchBase + waveRow) * 64 + nt * 16 + nL;
#pragma unroll
                    for (int r = 0; r < 8; ++r)
                        vr[(size_t)(r + hi * 8) * 64] = acc[r];
                } else {
                    __bf16* dst = (mat == 0) ? qs : ks;
#pragma unroll
                    for (int r = 0; r < 8; ++r)
                        dst[(wv * 16 + r + hi * 8) * 64 + nt * 16 + nL] = (__bf16)acc[r];
                }
            }
        }
        __syncthreads();

        // ---- attention statistics on this 128-row slab ----
        // Skq via WMMA: per wave, its 64 assigned rows = 2 K-chunks of 32.
        // A = K^T (d x r): a[t] from ks at row rA, col sh*16 + (L&15)
        // B = Q   (r x e): b[t] from qs at row rB, col sh*16 + (L&15)
#pragma unroll
        for (int kc2 = 0; kc2 < 2; ++kc2) {
            const int rbase = shalf * 64 + kc2 * 32;
            v16bf ka, qb;
#pragma unroll
            for (int t = 0; t < 16; ++t) {
                int rA = rbase + hi * 16 + t;
                ka[t] = ks[rA * 64 + sh * 16 + nL];
                int rB = rbase + hi * 16 + t;
                qb[t] = qs[rB * 64 + sh * 16 + nL];
            }
            skqAcc = __builtin_amdgcn_wmma_f32_16x16x32_bf16(
                         false, ka, false, qb, (short)0, skqAcc, false, false);
        }
        // Sq2/Sk2 diagonals (scalar, cheap)
        if (tid < 128) {
            const __bf16* src = (tid < 64) ? qs : ks;
            int c = tid & 63;
            float acc = 0.f;
            for (int r = 0; r < 128; ++r) {
                float v = (float)src[r * 64 + c];
                acc += v * v;
            }
            snorm += acc;
        }
        __syncthreads();
    }

    // partial layout per block: [2048 skq (wave*256 + d*16 + e)][64 Sq2][64 Sk2]
    float* pb = part + (size_t)blockIdx.x * 2176;
#pragma unroll
    for (int r = 0; r < 8; ++r)
        pb[wv * 256 + (r + hi * 8) * 16 + nL] = skqAcc[r];
    if (tid < 128) pb[2048 + tid] = snorm;
}

// ---------------------------------------------------------------------------
// Kernel 2: reduce partials -> l2-normalize (over tokens, eps) -> rescale ->
//           softmax over e. One block per (b,h); thread = one (d,e).
// ---------------------------------------------------------------------------
__global__ __launch_bounds__(256)
void k_attn(const float* __restrict__ part, const float* __restrict__ rescale,
            float* __restrict__ attn)
{
    __shared__ float sm[256];
    __shared__ float ex[256];
    __shared__ float nrm[32];    // [0..15]=||k_d||, [16..31]=||q_e||
    const int b = blockIdx.x >> 2;
    const int h = blockIdx.x & 3;
    const int tid = threadIdx.x;
    const int d = tid >> 4, e = tid & 15;

    const float* pbase = part + (size_t)b * 128 * 2176;
    float s = 0.f;
    for (int blk = 0; blk < 128; ++blk) {
        const float* pp = pbase + (size_t)blk * 2176;
        s += pp[h * 256 + tid] + pp[(h + 4) * 256 + tid];   // both row-halves
    }

    if (tid < 32) {
        int off = (tid < 16) ? (2112 + h * 16 + tid)          // Sk2[d]
                             : (2048 + h * 16 + (tid - 16));  // Sq2[e]
        float a = 0.f;
        for (int blk = 0; blk < 128; ++blk) a += pbase[blk * 2176 + off];
        nrm[tid] = fmaxf(sqrtf(a), 1e-12f);
    }
    __syncthreads();

    float val = s / (nrm[d] * nrm[16 + e]) * rescale[h];
    sm[tid] = val;
    __syncthreads();
    float m = -1e30f;
    for (int i = 0; i < 16; ++i) m = fmaxf(m, sm[d * 16 + i]);
    float eV = expf(val - m);
    ex[tid] = eV;
    __syncthreads();
    float sum = 0.f;
    for (int i = 0; i < 16; ++i) sum += ex[d * 16 + i];
    attn[(b * 4 + h) * 256 + tid] = eV / sum;
}

// ---------------------------------------------------------------------------
// Kernel 2b: fold attn into Wproj: Meff[he,col] = sum_d attn[h,d,e]*Wproj[hd,col]
// ---------------------------------------------------------------------------
__global__ __launch_bounds__(256)
void k_meff(const float* __restrict__ attn, const float* __restrict__ Wproj,
            float* __restrict__ meff)
{
    const int b = blockIdx.x;
    for (int i = threadIdx.x; i < 4096; i += 256) {
        int he = i >> 6; int h = he >> 4; int e = he & 15; int col = i & 63;
        const float* A = attn + (b * 4 + h) * 256;
        float acc = 0.f;
#pragma unroll
        for (int d2 = 0; d2 < 16; ++d2)
            acc += A[d2 * 16 + e] * Wproj[(h * 16 + d2) * 64 + col];
        meff[b * 4096 + i] = acc;        // [he][col]
    }
}

// ---------------------------------------------------------------------------
// Kernel 3: y = V @ Meff + bias  (streaming bf16-WMMA GEMM) -> d_out.
// Meff tile (16KB) is DMA'd into LDS by the Tensor Data Mover when available
// (D# per CDNA5 ISA; 1-D tile, data_size=4B, type=2).
// grid: 512 blocks (256/batch), 512 rows/block.
// ---------------------------------------------------------------------------
__global__ __launch_bounds__(256)
void k_out(const float* __restrict__ vbuf, const float* __restrict__ meff,
           const float* __restrict__ bias, float* __restrict__ out)
{
    __shared__ float  stage[4096];       // raw f32 Meff tile, 16KB
    __shared__ __bf16 Mt[64 * 64];       // [col][he] transposed bf16, 8KB
    const int tid = threadIdx.x;
    const int wv  = tid >> 5;
    const int L   = tid & 31;
    const int b   = blockIdx.x >> 8;
    const int blk = blockIdx.x & 255;
    const size_t batchBase = (size_t)b * NNs;
    const int rowBlockBase = blk * 512;

#if __has_builtin(__builtin_amdgcn_tensor_load_to_lds)
    if (tid < 32) {   // one wave issues the TDM descriptor (EXEC ignored)
        unsigned ldsOff = (unsigned)(size_t)&stage[0];          // LDS aperture low bits
        unsigned long long ga = (unsigned long long)(size_t)(meff + (size_t)b * 4096);
        v4u g0;
        g0[0] = 1u;                                             // count=1, user mode
        g0[1] = ldsOff;                                         // lds_addr
        g0[2] = (unsigned)(ga & 0xffffffffu);                   // global_addr[31:0]
        g0[3] = (unsigned)((ga >> 32) & 0x01ffffffu) | (2u << 30); // addr[56:32], type=2
        v8i g1;
        g1[0] = (int)(2u << 16);        // data_size=4B; no mask/iterate/pad
        g1[1] = (int)(4096u << 16);     // tensor_dim0 = 4096 (bits 79:48)
        g1[2] = (int)(1u << 16);        // tensor_dim1 = 1    (bits 111:80)
        g1[3] = (int)(4096u << 16);     // tile_dim0   = 4096 (bits 127:112)
        g1[4] = 0;                      // tile_dim1 = tile_dim2 = 0 (unused)
        g1[5] = 4096;                   // tensor_dim0_stride[31:0]
        g1[6] = 0;
        g1[7] = 0;
        v4i gz = {0, 0, 0, 0};
    #if __clang_major__ >= 23
        v8i gz8 = {0, 0, 0, 0, 0, 0, 0, 0};
        __builtin_amdgcn_tensor_load_to_lds(g0, g1, gz, gz, gz8, 0);
    #else
        __builtin_amdgcn_tensor_load_to_lds(g0, g1, gz, gz, 0);
    #endif
        __builtin_amdgcn_s_wait_tensorcnt(0);
    }
    __syncthreads();
#else
    for (int i = tid; i < 4096; i += 256) stage[i] = meff[(size_t)b * 4096 + i];
    __syncthreads();
#endif

    // transpose + convert: Mt[col][he] = stage[he][col]
    for (int i = tid; i < 4096; i += 256) {
        int col = i >> 6, k = i & 63;
        Mt[i] = (__bf16)stage[k * 64 + col];
    }
    __syncthreads();

    const int Mrow = L & 15;
    const int hi   = (L >> 4) & 1;
    const int nL   = L & 15;
    float bcol[4];
#pragma unroll
    for (int nt = 0; nt < 4; ++nt) bcol[nt] = bias[nt * 16 + nL];

    for (int slab = 0; slab < 4; ++slab) {
        const int waveRow = rowBlockBase + slab * 128 + wv * 16;
        const float* vr = vbuf + (batchBase + waveRow + Mrow) * 64;
        if (slab < 3)
            __builtin_prefetch(vbuf + (batchBase + waveRow + Mrow + 128) * 64, 0, 1);
        v16bf a0, a1;
#pragma unroll
        for (int t = 0; t < 16; ++t) {
            int k0 = (hi << 4) + t;
            a0[t] = (__bf16)vr[k0];
            a1[t] = (__bf16)vr[32 + k0];
        }
#pragma unroll
        for (int nt = 0; nt < 4; ++nt) {
            const __bf16* bp = &Mt[(nt * 16 + nL) * 64 + hi * 16];
            v16bf b0, b1;
#pragma unroll
            for (int t = 0; t < 16; ++t) { b0[t] = bp[t]; b1[t] = bp[32 + t]; }
            v8f acc = {};
            acc = __builtin_amdgcn_wmma_f32_16x16x32_bf16(
                      false, a0, false, b0, (short)0, acc, false, false);
            acc = __builtin_amdgcn_wmma_f32_16x16x32_bf16(
                      false, a1, false, b1, (short)0, acc, false, false);
            float* orow = out + (batchBase + waveRow) * 64 + nt * 16 + nL;
#pragma unroll
            for (int r = 0; r < 8; ++r)
                orow[(size_t)(r + hi * 8) * 64] = acc[r] + bcol[nt];
        }
    }
}

// ---------------------------------------------------------------------------
// Kernel 4: depthwise 3x3x3 conv (SAME, zero pad), optional exact GELU,
// optional accumulate. Filter bank (64x27 f32, 6912B) staged in LDS via
// async global->LDS b128 copies when available. One block per (b,z,y) line.
// ---------------------------------------------------------------------------
__device__ __forceinline__ float gelu_exact(float v) {
    return 0.5f * v * (1.0f + erff(v * 0.70710678118654752f));
}

__global__ __launch_bounds__(256)
void k_dwconv(const float* __restrict__ src, const float* __restrict__ w,
              float* __restrict__ dst, int doGelu, int accumulate)
{
    __shared__ float wl[64 * 27];      // 6912 bytes = 432 x b128
    const int tid = threadIdx.x;

#if __has_builtin(__builtin_amdgcn_global_load_async_to_lds_b128)
    for (int i = tid; i < 432; i += 256) {
        __builtin_amdgcn_global_load_async_to_lds_b128(
            (__attribute__((address_space(1))) v4i_vec*)(w + i * 4),
            (__attribute__((address_space(3))) v4i_vec*)(&wl[i * 4]), 0, 0);
    }
  #if __has_builtin(__builtin_amdgcn_s_wait_asynccnt)
    __builtin_amdgcn_s_wait_asynccnt(0);
  #else
    asm volatile("s_wait_asynccnt 0x0" ::: "memory");
  #endif
#else
    for (int i = tid; i < 64 * 27; i += 256) wl[i] = w[i];
#endif
    __syncthreads();

    int lin = blockIdx.x;
    const int y = lin % HHs; lin /= HHs;
    const int z = lin % DD;  lin /= DD;
    const int b = lin;
    const size_t base = (size_t)b * NNs * 64;

    for (int i = tid; i < WWs * CCs; i += 256) {
        const int xq = i >> 6;
        const int c  = i & 63;
        float acc = 0.f;
#pragma unroll
        for (int dz = -1; dz <= 1; ++dz) {
            const int zz = z + dz; if (zz < 0 || zz >= DD) continue;
#pragma unroll
            for (int dy = -1; dy <= 1; ++dy) {
                const int yy = y + dy; if (yy < 0 || yy >= HHs) continue;
#pragma unroll
                for (int dx = -1; dx <= 1; ++dx) {
                    const int xx = xq + dx; if (xx < 0 || xx >= WWs) continue;
                    const size_t n = ((size_t)zz * HHs + yy) * WWs + xx;
                    acc += src[base + n * 64 + c] *
                           wl[c * 27 + (dz + 1) * 9 + (dy + 1) * 3 + (dx + 1)];
                }
            }
        }
        if (doGelu) acc = gelu_exact(acc);
        const size_t oidx = base + (((size_t)z * HHs + y) * WWs + xq) * 64 + c;
        if (accumulate) dst[oidx] += acc; else dst[oidx] = acc;
    }
}

// ---------------------------------------------------------------------------
// Launch
// ---------------------------------------------------------------------------
extern "C" void kernel_launch(void* const* d_in, const int* in_sizes, int n_in,
                              void* d_out, int out_size, void* d_ws, size_t ws_size,
                              hipStream_t stream)
{
    const float* x   = (const float*)d_in[0];
    const float* Wq  = (const float*)d_in[1];
    const float* Wk  = (const float*)d_in[2];
    const float* Wv  = (const float*)d_in[3];
    const float* rs  = (const float*)d_in[4];
    const float* Wp  = (const float*)d_in[5];
    const float* bp  = (const float*)d_in[6];
    const float* Wc1 = (const float*)d_in[7];
    const float* Wc2 = (const float*)d_in[8];
    float* out = (float*)d_out;
    float* ws  = (float*)d_ws;

    // workspace layout (floats)
    float* vbuf = ws;                          // 16,777,216 (64 MB)
    float* p1   = ws + (size_t)16777216;       // 16,777,216 (64 MB)
    float* part = ws + (size_t)2 * 16777216;   // 256 * 2176 = 557,056
    float* attn = part + 557056;               // 2,048
    float* meff = attn + 2048;                 // 8,192

    k_qkv <<<256, 256, 0, stream>>>(x, Wq, Wk, Wv, vbuf, part);
    k_attn<<<Bq * NHEAD, 256, 0, stream>>>(part, rs, attn);
    k_meff<<<Bq, 256, 0, stream>>>(attn, Wp, meff);
    k_out <<<512, 256, 0, stream>>>(vbuf, meff, bp, out);
    k_dwconv<<<Bq * DD * HHs, 256, 0, stream>>>(vbuf, Wc1, p1, 1, 0);
    k_dwconv<<<Bq * DD * HHs, 256, 0, stream>>>(p1, Wc2, out, 0, 1);
}
